// GNNProcessor_17583596110613
// MI455X (gfx1250) — matrix operand order repaired and checked by reading, not compiled, not run on hardware
//
#include <hip/hip_runtime.h>
#include <hip/hip_bf16.h>
#include <stdint.h>

// ---------------------------------------------------------------------------
// GNN Transformer layer for MI455X (gfx1250), wave32 + WMMA bf16.
// Register-blocked GEMMs: one wave holds 4-8 accumulator tiles so each A
// fragment feeds 4-8 back-to-back v_wmma ops.
// ---------------------------------------------------------------------------

#define NNODES 50000
#define NEDGES 600000
#define DIM    128
#define HEADS  16
#define HIDDIM 512

typedef __attribute__((ext_vector_type(16))) __bf16 v16bf;
typedef __attribute__((ext_vector_type(8)))  __bf16 v8bf;
typedef __attribute__((ext_vector_type(8)))  float  v8f;
typedef __attribute__((ext_vector_type(4)))  float  v4f;

static __device__ __forceinline__ v8f vzero8() {
  v8f z;
#pragma unroll
  for (int i = 0; i < 8; ++i) z[i] = 0.0f;
  return z;
}

static __device__ __forceinline__ v8f wmma_bf16(v16bf a, v16bf b, v8f c) {
  // D = A(16x32 bf16) * B(32x16 bf16) + C(16x16 f32)
  return __builtin_amdgcn_wmma_f32_16x16x32_bf16(false, a, false, b, (short)0, c,
                                                 false, false);
}

// ---- Fragment loaders matching CDNA5 ISA 7.12.2 layouts (wave32) -----------
// A 16x32 bf16: lanes 0-15 -> M=lane, K in {k0..k0+7, k0+16..k0+23}
//               lanes 16-31 -> M=lane-16, K in {k0+8..k0+15, k0+24..k0+31}
static __device__ __forceinline__ v16bf load_a_bf16(const __bf16* A, int lda,
                                                    int m0, int k0, int lane) {
  int m  = m0 + (lane & 15);
  int hi = (lane >> 4) & 1;
  const __bf16* p = A + (size_t)m * lda + k0 + hi * 8;
  union { v16bf v; v8bf h[2]; } u;
  u.h[0] = *(const v8bf*)(p);        // 16B aligned
  u.h[1] = *(const v8bf*)(p + 16);
  return u.v;
}

static __device__ __forceinline__ v16bf load_a_f32_cvt(const float* A, int lda,
                                                       int m0, int k0, int lane) {
  int m  = m0 + (lane & 15);
  int hi = (lane >> 4) & 1;
  const float* p = A + (size_t)m * lda + k0 + hi * 8;
  v4f f0 = *(const v4f*)(p);
  v4f f1 = *(const v4f*)(p + 4);
  v4f f2 = *(const v4f*)(p + 16);
  v4f f3 = *(const v4f*)(p + 20);
  v16bf r;
#pragma unroll
  for (int j = 0; j < 4; ++j) {
    r[j]      = (__bf16)f0[j];
    r[4 + j]  = (__bf16)f1[j];
    r[8 + j]  = (__bf16)f2[j];
    r[12 + j] = (__bf16)f3[j];
  }
  return r;
}

// A = A1 + A2 (both f32), converted to bf16
static __device__ __forceinline__ v16bf load_a_sum_cvt(const float* A1, int lda1,
                                                       const float* A2, int lda2,
                                                       int m0, int k0, int lane) {
  int m  = m0 + (lane & 15);
  int hi = (lane >> 4) & 1;
  const float* p1 = A1 + (size_t)m * lda1 + k0 + hi * 8;
  const float* p2 = A2 + (size_t)m * lda2 + k0 + hi * 8;
  v16bf r;
#pragma unroll
  for (int j = 0; j < 8; ++j) {
    r[j]     = (__bf16)(p1[j] + p2[j]);
    r[8 + j] = (__bf16)(p1[16 + j] + p2[16 + j]);
  }
  return r;
}

// B 32x16 bf16 from output-channel-major weights Bt[n][k] (ld = K):
// lane n = lane&15, K in {k0+8h..k0+8h+7, k0+16+8h..k0+16+8h+7}, h = lane>>4.
static __device__ __forceinline__ v16bf load_b(const __bf16* Bt, int ldb,
                                               int n0, int k0, int lane) {
  int n  = n0 + (lane & 15);
  int kg = (lane >> 4) & 1;
  const __bf16* p = Bt + (size_t)n * ldb + k0 + kg * 8;
  union { v16bf v; v8bf h[2]; } u;
  u.h[0] = *(const v8bf*)(p);
  u.h[1] = *(const v8bf*)(p + 16);
  return u.v;
}

// ---------------------------------------------------------------------------
// Kernel 0: convert/transposes weights to bf16 n-major, concat qkvs bias
// ---------------------------------------------------------------------------
__global__ void convert_weights_kernel(const float* Wq, const float* Wk,
                                       const float* Wv, const float* Ws,
                                       const float* We, const float* Wp,
                                       const float* W1, const float* W2,
                                       const float* bq, const float* bk,
                                       const float* bv, const float* bs,
                                       __bf16* wqkvs_t, __bf16* we_t, __bf16* wp_t,
                                       __bf16* w1_t, __bf16* w2_t, float* bias_qkvs) {
  int idx = blockIdx.x * blockDim.x + threadIdx.x;
  if (idx < 65536) {  // Wq|Wk|Wv|Ws -> wqkvs_t[512][128]
    int nout = idx >> 7, k = idx & 127;
    int g = nout >> 7, n = nout & 127;
    const float* W = (g == 0) ? Wq : (g == 1) ? Wk : (g == 2) ? Wv : Ws;
    wqkvs_t[(size_t)nout * 128 + k] = (__bf16)W[(size_t)k * 128 + n];
    return;
  }
  idx -= 65536;
  if (idx < 16384) {  // We -> we_t[128][128]
    int n = idx >> 7, k = idx & 127;
    we_t[(size_t)n * 128 + k] = (__bf16)We[(size_t)k * 128 + n];
    return;
  }
  idx -= 16384;
  if (idx < 16384) {  // Wp -> wp_t[128][128]
    int n = idx >> 7, k = idx & 127;
    wp_t[(size_t)n * 128 + k] = (__bf16)Wp[(size_t)k * 128 + n];
    return;
  }
  idx -= 16384;
  if (idx < 65536) {  // W1[128][512] -> w1_t[512][128]
    int n = idx >> 7, k = idx & 127;
    w1_t[(size_t)n * 128 + k] = (__bf16)W1[(size_t)k * 512 + n];
    return;
  }
  idx -= 65536;
  if (idx < 65536) {  // W2[512][128] -> w2_t[128][512]
    int n = idx >> 9, k = idx & 511;
    w2_t[(size_t)n * 512 + k] = (__bf16)W2[(size_t)k * 128 + n];
    return;
  }
  idx -= 65536;
  if (idx < 512) {    // concat biases
    int g = idx >> 7, c = idx & 127;
    const float* B = (g == 0) ? bq : (g == 1) ? bk : (g == 2) ? bv : bs;
    bias_qkvs[idx] = B[c];
  }
}

// ---------------------------------------------------------------------------
// LayerNorm (row of 128) f32 -> bf16
// ---------------------------------------------------------------------------
__global__ void ln_to_bf16_kernel(const float* __restrict__ in,
                                  const float* __restrict__ g,
                                  const float* __restrict__ b,
                                  __bf16* __restrict__ out) {
  int row = blockIdx.x;
  int c = threadIdx.x;  // 0..127
  float v = in[(size_t)row * 128 + c];
  __shared__ float s1[128], s2[128];
  s1[c] = v;
  s2[c] = v * v;
  __syncthreads();
  for (int off = 64; off > 0; off >>= 1) {
    if (c < off) { s1[c] += s1[c + off]; s2[c] += s2[c + off]; }
    __syncthreads();
  }
  float mean = s1[0] * (1.0f / 128.0f);
  float var  = s2[0] * (1.0f / 128.0f) - mean * mean;
  float rstd = rsqrtf(var + 1e-5f);
  out[(size_t)row * 128 + c] = (__bf16)((v - mean) * rstd * g[c] + b[c]);
}

// ---------------------------------------------------------------------------
// GEMM: qkvs = xn[N,128] @ Wqkvs_t^T + bias  -> [N,512] f32
// one wave = 16x64 (4 acc tiles); A loaded once per K-step
// ---------------------------------------------------------------------------
__global__ void gemm_qkvs_kernel(const __bf16* __restrict__ xn,
                                 const __bf16* __restrict__ Wt,
                                 const float* __restrict__ bias,
                                 float* __restrict__ out) {
  int wid  = blockIdx.x * (blockDim.x >> 5) + (threadIdx.x >> 5);
  int lane = threadIdx.x & 31;
  const int TILES = (NNODES / 16) * (HIDDIM / 64);   // 3125 * 8
  if (wid >= TILES) return;
  int mt = wid >> 3, ng = wid & 7;
  int m0 = mt * 16, n0 = ng * 64;
  v8f acc[4];
#pragma unroll
  for (int t = 0; t < 4; ++t) acc[t] = vzero8();
#pragma unroll
  for (int kk = 0; kk < 128; kk += 32) {
    v16bf a = load_a_bf16(xn, 128, m0, kk, lane);
#pragma unroll
    for (int t = 0; t < 4; ++t)
      acc[t] = wmma_bf16(a, load_b(Wt, 128, n0 + t * 16, kk, lane), acc[t]);
  }
  int hi = lane >> 4;
#pragma unroll
  for (int t = 0; t < 4; ++t) {
    int n = n0 + t * 16 + (lane & 15);
    float bn = bias[n];
#pragma unroll
    for (int r = 0; r < 8; ++r) {
      int m = m0 + r + 8 * hi;
      out[(size_t)m * 512 + n] = acc[t][r] + bn;
    }
  }
}

// ---------------------------------------------------------------------------
// Fused edge kernel: e = edge_attr @ We + be via WMMA (16 edges/wave),
// alpha[e,h] = sum_c q[dst,h,c]*(k[src,h,c]+e[h,c]) / sqrt(8),
// segment-max via monotonic-u32 atomicMax.
// qkvs rows: [q(0:128) | k(128:256) | v(256:384) | s(384:512)]
// ---------------------------------------------------------------------------
__global__ void edge_alpha_kernel(const float* __restrict__ edge_attr,
                                  const __bf16* __restrict__ we_t,
                                  const float* __restrict__ be,
                                  const int* __restrict__ ei,
                                  const float* __restrict__ qkvs,
                                  float* __restrict__ alpha,
                                  unsigned int* __restrict__ amax) {
  int wid  = blockIdx.x * (blockDim.x >> 5) + (threadIdx.x >> 5);
  int lane = threadIdx.x & 31;
  int m0 = wid * 16;
  if (m0 >= NEDGES) return;  // wave-uniform guard (EXEC stays all-ones)

  // prefetch next edge tile's rows
  if (m0 + 16 < NEDGES)
    __builtin_prefetch(edge_attr + ((size_t)(m0 + 16) + (lane & 15)) * 128, 0, 1);

  v8f acc[8];
#pragma unroll
  for (int t = 0; t < 8; ++t) acc[t] = vzero8();

#pragma unroll
  for (int kk = 0; kk < 128; kk += 32) {
    v16bf a = load_a_f32_cvt(edge_attr, 128, m0, kk, lane);
#pragma unroll
    for (int t = 0; t < 8; ++t)
      acc[t] = wmma_bf16(a, load_b(we_t, 128, t * 16, kk, lane), acc[t]);
  }

  int nl = lane & 15;
  int hi = lane >> 4;
  int srcs[8], dsts[8];
#pragma unroll
  for (int r = 0; r < 8; ++r) {
    int e = m0 + r + 8 * hi;
    srcs[r] = ei[e];
    dsts[r] = ei[NEDGES + e];
  }
  const float scale = 0.35355339059327373f;  // 1/sqrt(8)
  const float* qbase = qkvs;          // q at col 0
  const float* kbase = qkvs + 128;    // k at col 128

#pragma unroll
  for (int t = 0; t < 8; ++t) {
    int n = t * 16 + nl;              // channel index 0..127
    float bias = be[n];
#pragma unroll
    for (int r = 0; r < 8; ++r) {
      float ke   = kbase[(size_t)srcs[r] * 512 + n] + acc[t][r] + bias;
      float prod = qbase[(size_t)dsts[r] * 512 + n] * ke;
      prod += __shfl_xor(prod, 1, 32);
      prod += __shfl_xor(prod, 2, 32);
      prod += __shfl_xor(prod, 4, 32);
      if ((lane & 7) == 0) {
        int e = m0 + r + 8 * hi;
        int h = n >> 3;               // head = 2t + ((lane>>3)&1)
        float aval = prod * scale;
        alpha[(size_t)e * 16 + h] = aval;
        unsigned int bits = __float_as_uint(aval);
        unsigned int mono = (bits & 0x80000000u) ? ~bits : (bits | 0x80000000u);
        atomicMax(&amax[(size_t)dsts[r] * 16 + h], mono);
      }
    }
  }
}

// ---------------------------------------------------------------------------
// ex = exp(alpha - amax[dst]); denom += ex  (alpha overwritten with ex)
// ---------------------------------------------------------------------------
__global__ void softmax_exp_kernel(float* __restrict__ alpha,
                                   const unsigned int* __restrict__ amax,
                                   float* __restrict__ denom,
                                   const int* __restrict__ ei) {
  int idx = blockIdx.x * blockDim.x + threadIdx.x;
  if (idx >= NEDGES * 16) return;
  int e = idx >> 4, h = idx & 15;
  int d = ei[NEDGES + e];
  unsigned int mono = amax[(size_t)d * 16 + h];
  unsigned int bits = (mono & 0x80000000u) ? (mono & 0x7FFFFFFFu) : ~mono;
  float mx = __uint_as_float(bits);
  float ex = __expf(alpha[idx] - mx);
  alpha[idx] = ex;
  atomicAdd(&denom[(size_t)d * 16 + h], ex);
}

// ---------------------------------------------------------------------------
// agg[dst] += v[src] * attn   (32 threads per edge, 4 channels each)
// ---------------------------------------------------------------------------
__global__ void aggregate_kernel(const float* __restrict__ alpha,  // holds ex
                                 const float* __restrict__ denom,
                                 const float* __restrict__ qkvs,   // v at +256
                                 const int* __restrict__ ei,
                                 float* __restrict__ agg) {
  int gid = blockIdx.x * blockDim.x + threadIdx.x;
  int e = gid >> 5;
  if (e >= NEDGES) return;
  int t = gid & 31;
  int ch0 = t * 4;
  int h = ch0 >> 3;
  int s = ei[e], d = ei[NEDGES + e];
  float attn = alpha[(size_t)e * 16 + h] / denom[(size_t)d * 16 + h];
  v4f vv = *(const v4f*)(qkvs + (size_t)s * 512 + 256 + ch0);
  float* out = agg + (size_t)d * 128 + ch0;
  atomicAdd(&out[0], vv[0] * attn);
  atomicAdd(&out[1], vv[1] * attn);
  atomicAdd(&out[2], vv[2] * attn);
  atomicAdd(&out[3], vv[3] * attn);
}

// ---------------------------------------------------------------------------
// out2 = (agg + x_r) @ Wp + bp + x    (x_r = qkvs cols 384:512)
// one wave = 16x128 (8 acc tiles); fused A built once per K-step
// ---------------------------------------------------------------------------
__global__ void gemm_p_kernel(const float* __restrict__ agg,
                              const float* __restrict__ qkvs,
                              const __bf16* __restrict__ wp_t,
                              const float* __restrict__ bp,
                              const float* __restrict__ x,
                              float* __restrict__ out2) {
  int wid  = blockIdx.x * (blockDim.x >> 5) + (threadIdx.x >> 5);
  int lane = threadIdx.x & 31;
  const int TILES = NNODES / 16;   // 3125 row-tiles, full 128 cols per wave
  if (wid >= TILES) return;
  int m0 = wid * 16;
  v8f acc[8];
#pragma unroll
  for (int t = 0; t < 8; ++t) acc[t] = vzero8();
#pragma unroll
  for (int kk = 0; kk < 128; kk += 32) {
    v16bf a = load_a_sum_cvt(agg, 128, qkvs + 384, 512, m0, kk, lane);
#pragma unroll
    for (int t = 0; t < 8; ++t)
      acc[t] = wmma_bf16(a, load_b(wp_t, 128, t * 16, kk, lane), acc[t]);
  }
  int hi = lane >> 4;
#pragma unroll
  for (int t = 0; t < 8; ++t) {
    int n = t * 16 + (lane & 15);
    float bn = bp[n];
#pragma unroll
    for (int r = 0; r < 8; ++r) {
      int m = m0 + r + 8 * hi;
      out2[(size_t)m * 128 + n] = acc[t][r] + bn + x[(size_t)m * 128 + n];
    }
  }
}

// ---------------------------------------------------------------------------
// hid = silu(hb @ W1 + b1)  -> bf16 [N,512]
// one wave = 16x64 (4 acc tiles)
// ---------------------------------------------------------------------------
__global__ void gemm_mlp1_kernel(const __bf16* __restrict__ hb,
                                 const __bf16* __restrict__ w1_t,
                                 const float* __restrict__ b1,
                                 __bf16* __restrict__ hid) {
  int wid  = blockIdx.x * (blockDim.x >> 5) + (threadIdx.x >> 5);
  int lane = threadIdx.x & 31;
  const int TILES = (NNODES / 16) * (HIDDIM / 64);   // 3125 * 8
  if (wid >= TILES) return;
  int mt = wid >> 3, ng = wid & 7;
  int m0 = mt * 16, n0 = ng * 64;
  v8f acc[4];
#pragma unroll
  for (int t = 0; t < 4; ++t) acc[t] = vzero8();
#pragma unroll
  for (int kk = 0; kk < 128; kk += 32) {
    v16bf a = load_a_bf16(hb, 128, m0, kk, lane);
#pragma unroll
    for (int t = 0; t < 4; ++t)
      acc[t] = wmma_bf16(a, load_b(w1_t, 128, n0 + t * 16, kk, lane), acc[t]);
  }
  int hi = lane >> 4;
#pragma unroll
  for (int t = 0; t < 4; ++t) {
    int n = n0 + t * 16 + (lane & 15);
    float bn = b1[n];
#pragma unroll
    for (int r = 0; r < 8; ++r) {
      int m = m0 + r + 8 * hi;
      float val = acc[t][r] + bn;
      float s = val / (1.0f + __expf(-val));  // silu
      hid[(size_t)m * 512 + n] = (__bf16)s;
    }
  }
}

// ---------------------------------------------------------------------------
// d_out = hid @ W2 + b2 + out2   (K = 512)
// one wave = 16x128 (8 acc tiles); 16 A loads vs 128 WMMAs
// ---------------------------------------------------------------------------
__global__ void gemm_mlp2_kernel(const __bf16* __restrict__ hid,
                                 const __bf16* __restrict__ w2_t,
                                 const float* __restrict__ b2,
                                 const float* __restrict__ out2,
                                 float* __restrict__ out) {
  int wid  = blockIdx.x * (blockDim.x >> 5) + (threadIdx.x >> 5);
  int lane = threadIdx.x & 31;
  const int TILES = NNODES / 16;   // 3125 row-tiles, full 128 cols per wave
  if (wid >= TILES) return;
  int m0 = wid * 16;
  v8f acc[8];
#pragma unroll
  for (int t = 0; t < 8; ++t) acc[t] = vzero8();
  for (int kk = 0; kk < 512; kk += 32) {
    v16bf a = load_a_bf16(hid, 512, m0, kk, lane);
#pragma unroll
    for (int t = 0; t < 8; ++t)
      acc[t] = wmma_bf16(a, load_b(w2_t, 512, t * 16, kk, lane), acc[t]);
  }
  int hi = lane >> 4;
#pragma unroll
  for (int t = 0; t < 8; ++t) {
    int n = t * 16 + (lane & 15);
    float bn = b2[n];
#pragma unroll
    for (int r = 0; r < 8; ++r) {
      int m = m0 + r + 8 * hi;
      out[(size_t)m * 128 + n] = acc[t][r] + bn + out2[(size_t)m * 128 + n];
    }
  }
}

// ---------------------------------------------------------------------------
// Launch
// ---------------------------------------------------------------------------
extern "C" void kernel_launch(void* const* d_in, const int* in_sizes, int n_in,
                              void* d_out, int out_size, void* d_ws, size_t ws_size,
                              hipStream_t stream) {
  (void)in_sizes; (void)n_in; (void)out_size; (void)ws_size;

  const float* x     = (const float*)d_in[0];
  const int*   ei    = (const int*)  d_in[1];
  const float* eattr = (const float*)d_in[2];
  const float* ln1_g = (const float*)d_in[3];
  const float* ln1_b = (const float*)d_in[4];
  const float* Wq = (const float*)d_in[5];  const float* bq = (const float*)d_in[6];
  const float* Wk = (const float*)d_in[7];  const float* bk = (const float*)d_in[8];
  const float* Wv = (const float*)d_in[9];  const float* bv = (const float*)d_in[10];
  const float* Ws = (const float*)d_in[11]; const float* bs = (const float*)d_in[12];
  const float* We = (const float*)d_in[13]; const float* be = (const float*)d_in[14];
  const float* Wp = (const float*)d_in[15]; const float* bp = (const float*)d_in[16];
  const float* mlp_g = (const float*)d_in[17]; const float* mlp_b = (const float*)d_in[18];
  const float* W1 = (const float*)d_in[19]; const float* b1 = (const float*)d_in[20];
  const float* W2 = (const float*)d_in[21]; const float* b2 = (const float*)d_in[22];

  // Workspace layout (all offsets are multiples of 256 bytes)
  constexpr size_t OFF_WQKVS = 0;                                        // 512*128 bf16
  constexpr size_t OFF_WE    = OFF_WQKVS + (size_t)512 * 128 * 2;        // 128*128 bf16
  constexpr size_t OFF_WPT   = OFF_WE    + (size_t)128 * 128 * 2;        // 128*128 bf16
  constexpr size_t OFF_W1T   = OFF_WPT   + (size_t)128 * 128 * 2;        // 512*128 bf16
  constexpr size_t OFF_W2T   = OFF_W1T   + (size_t)512 * 128 * 2;        // 128*512 bf16
  constexpr size_t OFF_BQKVS = OFF_W2T   + (size_t)128 * 512 * 2;        // 512 f32
  constexpr size_t OFF_XN    = OFF_BQKVS + (size_t)512 * 4;              // N*128 bf16
  constexpr size_t OFF_QKVS  = OFF_XN    + (size_t)NNODES * 128 * 2;     // N*512 f32
  constexpr size_t OFF_ALPHA = OFF_QKVS  + (size_t)NNODES * 512 * 4;     // E*16 f32
  constexpr size_t OFF_AMAX  = OFF_ALPHA + (size_t)NEDGES * 16 * 4;      // N*16 u32
  constexpr size_t OFF_DENOM = OFF_AMAX  + (size_t)NNODES * 16 * 4;      // N*16 f32
  constexpr size_t OFF_AGG   = OFF_DENOM + (size_t)NNODES * 16 * 4;      // N*128 f32
  constexpr size_t OFF_OUT2  = OFF_AGG   + (size_t)NNODES * 128 * 4;     // N*128 f32
  constexpr size_t OFF_HB    = OFF_OUT2  + (size_t)NNODES * 128 * 4;     // N*128 bf16
  constexpr size_t OFF_HID   = OFF_HB    + (size_t)NNODES * 128 * 2;     // N*512 bf16

  char* ws = (char*)d_ws;
  __bf16* wqkvs_t = (__bf16*)(ws + OFF_WQKVS);
  __bf16* we_t    = (__bf16*)(ws + OFF_WE);
  __bf16* wp_t    = (__bf16*)(ws + OFF_WPT);
  __bf16* w1_t    = (__bf16*)(ws + OFF_W1T);
  __bf16* w2_t    = (__bf16*)(ws + OFF_W2T);
  float*  bqkvs   = (float*) (ws + OFF_BQKVS);
  __bf16* xn_bf   = (__bf16*)(ws + OFF_XN);
  float*  qkvs    = (float*) (ws + OFF_QKVS);
  float*  alpha   = (float*) (ws + OFF_ALPHA);
  unsigned int* amax = (unsigned int*)(ws + OFF_AMAX);
  float*  denom   = (float*) (ws + OFF_DENOM);
  float*  agg     = (float*) (ws + OFF_AGG);
  float*  out2    = (float*) (ws + OFF_OUT2);
  __bf16* hb      = (__bf16*)(ws + OFF_HB);
  __bf16* hid     = (__bf16*)(ws + OFF_HID);

  // 0) weight pre-processing
  convert_weights_kernel<<<(229888 + 255) / 256, 256, 0, stream>>>(
      Wq, Wk, Wv, Ws, We, Wp, W1, W2, bq, bk, bv, bs,
      wqkvs_t, we_t, wp_t, w1_t, w2_t, bqkvs);

  // 1) xn = LN(x)
  ln_to_bf16_kernel<<<NNODES, 128, 0, stream>>>(x, ln1_g, ln1_b, xn_bf);

  // 2) qkvs = xn @ [Wq|Wk|Wv|Ws] + bias   (16x64 per wave)
  {
    int tiles = (NNODES / 16) * (HIDDIM / 64);       // 25000
    gemm_qkvs_kernel<<<(tiles + 7) / 8, 256, 0, stream>>>(xn_bf, wqkvs_t, bqkvs, qkvs);
  }

  // zero amax / denom / agg (contiguous region, capture-safe)
  hipMemsetAsync(ws + OFF_AMAX, 0,
                 (size_t)NNODES * 16 * 4 * 2 + (size_t)NNODES * 128 * 4, stream);

  // 3) fused edge GEMM + attention logits + segment max
  {
    int wave_tiles = NEDGES / 16;            // 37500
    int blocks = (wave_tiles + 7) / 8;       // 8 waves/block
    edge_alpha_kernel<<<blocks, 256, 0, stream>>>(eattr, we_t, be, ei, qkvs,
                                                  alpha, amax);
  }

  // 4) exp + segment sum
  softmax_exp_kernel<<<(NEDGES * 16) / 256, 256, 0, stream>>>(alpha, amax, denom, ei);

  // 5) scatter-aggregate weighted values
  aggregate_kernel<<<(NEDGES * 32) / 256, 256, 0, stream>>>(alpha, denom, qkvs, ei, agg);

  // 6) out2 = (agg + x_r) @ Wp + bp + x   (16x128 per wave)
  {
    int tiles = NNODES / 16;                         // 3125
    gemm_p_kernel<<<(tiles + 7) / 8, 256, 0, stream>>>(agg, qkvs, wp_t, bp, x, out2);
  }

  // 7) hb = LN(out2)
  ln_to_bf16_kernel<<<NNODES, 128, 0, stream>>>(out2, mlp_g, mlp_b, hb);

  // 8) hid = silu(hb @ W1 + b1)   (16x64 per wave)
  {
    int tiles = (NNODES / 16) * (HIDDIM / 64);       // 25000
    gemm_mlp1_kernel<<<(tiles + 7) / 8, 256, 0, stream>>>(hb, w1_t, b1, hid);
  }

  // 9) d_out = hid @ W2 + b2 + out2   (16x128 per wave, K=512)
  {
    int tiles = NNODES / 16;                         // 3125
    gemm_mlp2_kernel<<<(tiles + 7) / 8, 256, 0, stream>>>(hid, w2_t, b2, out2,
                                                          (float*)d_out);
  }
}
